// PointNet2Layer_20899310862372
// MI455X (gfx1250) — matrix operand order repaired and checked by reading, not compiled, not run on hardware
//
#include <hip/hip_runtime.h>

typedef __attribute__((ext_vector_type(16))) _Float16 v16h;
typedef __attribute__((ext_vector_type(8)))  _Float16 v8h;
typedef __attribute__((ext_vector_type(8)))  float    v8f;

namespace {
constexpr int kB = 16, kM = 4096, kK = 16, kCin = 64, kCf = 96, kCout = 128;
constexpr int kPts = kB * kM;                       // 65536 points
constexpr float kEps = 1e-5f;

// workspace byte offsets (all 16B aligned)
constexpr size_t OFF_IDX  = 0;                                    // int[kPts*kK]
constexpr size_t OFF_XH   = OFF_IDX + (size_t)kPts * kK * 4;      // f16[kPts*kCin]
constexpr size_t OFF_W1T  = OFF_XH  + (size_t)kPts * kCin * 2;    // f16[64][96]  (N-major)
constexpr size_t OFF_W2T  = OFF_W1T + (size_t)64 * 96 * 2;        // f16[128][64] (N-major)
constexpr size_t OFF_STAT = OFF_W2T + (size_t)128 * 64 * 2;       // float[256]: sum,sq,scale,shift
} // namespace

// ---------------------------------------------------------------- helpers ----
__device__ inline v8f vzero8() {
  v8f z;
  #pragma unroll
  for (int e = 0; e < 8; ++e) z[e] = 0.f;
  return z;
}

// Async global->LDS copy of 32 bytes per lane (2x B128), tracked by ASYNCcnt.
// (INST_OFFSET applies to BOTH addresses, so both ops share the same VGPRs.)
__device__ inline void async_copy_b256(void* lds_dst, const void* gsrc) {
  unsigned lds = (unsigned)(__SIZE_TYPE__)lds_dst;   // low 32 bits = LDS byte addr
  unsigned long long ga = (unsigned long long)gsrc;
  asm volatile("global_load_async_to_lds_b128 %0, %1, off"
               :: "v"(lds), "v"(ga) : "memory");
  asm volatile("global_load_async_to_lds_b128 %0, %1, off offset:16"
               :: "v"(lds), "v"(ga) : "memory");
}

__device__ inline void wait_asynccnt0() {
  asm volatile("s_wait_asynccnt 0" ::: "memory");
}

// A fragment, 16-bit 16x32 layout: lane half h, row M=lane%16.
// elements 0..7  -> K = kb*32 + h*8 + 0..7
// elements 8..15 -> K = kb*32 + 16 + h*8 + 0..7
__device__ inline v16h ld_fragA(const _Float16* row, int kb, int hx) {
  v8h lo = *(const v8h*)(row + kb * 32 + hx * 8);
  v8h hi = *(const v8h*)(row + kb * 32 + 16 + hx * 8);
  v16h r;
  #pragma unroll
  for (int e = 0; e < 8; ++e) { r[e] = lo[e]; r[8 + e] = hi[e]; }
  return r;
}

// B fragment, 16-bit 32x16 layout: lane half h, col N=lane%16.
// elements 0..7 -> K = kb*32 + h*8 + 0..7, elements 8..15 -> K = kb*32 + 16 + h*8 + 0..7
__device__ inline v16h ld_fragB(const _Float16* row, int kb, int hx) {
  const _Float16* base = row + kb * 32 + hx * 8;
  v8h lo = *(const v8h*)(base);
  v8h hi = *(const v8h*)(base + 16);
  v16h r;
  #pragma unroll
  for (int e = 0; e < 8; ++e) { r[e] = lo[e]; r[8 + e] = hi[e]; }
  return r;
}

// ------------------------------------------------------------ prep kernels ---
__global__ void prep_x_kernel(const float* __restrict__ x, _Float16* __restrict__ xh) {
  int t = blockIdx.x * blockDim.x + threadIdx.x;   // exactly kPts*kCin threads
  xh[t] = (_Float16)x[t];
}

__global__ void prep_w_kernel(const float* __restrict__ w1, const float* __restrict__ w2,
                              _Float16* __restrict__ w1t, _Float16* __restrict__ w2t,
                              float* __restrict__ stat) {
  int t = blockIdx.x * blockDim.x + threadIdx.x;
  if (t < 64 * 96) {
    // Reordered K-space: K 0..63 = features (w1 rows 3..66), K 64..66 = dxyz, 67..95 = 0
    int n = t / 96, k = t % 96;
    float v = 0.f;
    if (k < 64) v = w1[(k + 3) * 64 + n];
    else if (k < 67) v = w1[(k - 64) * 64 + n];
    w1t[n * 96 + k] = (_Float16)v;
  } else if (t < 64 * 96 + 128 * 64) {
    int u = t - 64 * 96;
    int n = u / 64, k = u % 64;
    w2t[n * 64 + k] = (_Float16)w2[k * 128 + n];
  } else if (t < 64 * 96 + 128 * 64 + 128) {
    stat[t - (64 * 96 + 128 * 64)] = 0.f;          // zero Σ and Σ² accumulators
  }
}

// ------------------------------------------------------------------- KNN -----
__global__ __launch_bounds__(256) void knn_kernel(const float* __restrict__ p,
                                                  int* __restrict__ idx) {
  __shared__ float sp[kM * 3];                      // 48 KB: whole cloud in LDS
  const int b = blockIdx.y;
  const float* pc = p + (size_t)b * kM * 3;
  for (int t = threadIdx.x; t < kM * 3; t += 256) sp[t] = pc[t];
  __syncthreads();

  const int i = blockIdx.x * 256 + threadIdx.x;
  const float px = sp[i * 3 + 0], py = sp[i * 3 + 1], pz = sp[i * 3 + 2];
  float bd[kK]; int bj[kK];
  #pragma unroll
  for (int t = 0; t < kK; ++t) { bd[t] = 3.4e38f; bj[t] = 0; }

  for (int j = 0; j < kM; ++j) {
    float dx = sp[j * 3 + 0] - px;
    float dy = sp[j * 3 + 1] - py;
    float dz = sp[j * 3 + 2] - pz;
    float d2 = dx * dx + dy * dy + dz * dz;
    if (d2 < bd[kK - 1]) {
      bd[kK - 1] = d2; bj[kK - 1] = j;
      #pragma unroll
      for (int t = kK - 1; t > 0; --t) {            // one bubble pass keeps it sorted
        if (bd[t] < bd[t - 1]) {
          float td = bd[t]; bd[t] = bd[t - 1]; bd[t - 1] = td;
          int   tj = bj[t]; bj[t] = bj[t - 1]; bj[t - 1] = tj;
        }
      }
    }
  }
  int* o = idx + (size_t)(b * kM + i) * kK;
  #pragma unroll
  for (int t = 0; t < kK; ++t) o[t] = bj[t];
}

// --------------------------------------------- shared stage + GEMM1 (WMMA) ---
// One wave handles one point: async-DMA its 16 neighbor rows (16x96 f16) into
// wave-private LDS, then D(16x64) = A(16x96) * w1t via 3x4 v_wmma_f32_16x16x32_f16.
__device__ inline void stage_and_gemm1(const float* __restrict__ p,
                                       const _Float16* __restrict__ xh,
                                       const int* __restrict__ idx,
                                       const _Float16* __restrict__ w1t,
                                       _Float16* frow, int point, int lane,
                                       v8f acc[4]) {
  const int b = point >> 12, i = point & (kM - 1);
  const int r = lane & 15, part = lane >> 4;
  const int j = idx[(size_t)point * kK + r];

  // 16 contiguous f16 features (32 bytes) per lane, moved global->LDS with no VGPR roundtrip
  async_copy_b256(frow + r * kCf + part * 32,
                  xh + ((size_t)(b * kM + j)) * kCin + part * 32);

  if (part == 0) {                                  // relative xyz + zero pad
    const float* pi = p + (size_t)(b * kM + i) * 3;
    const float* pj = p + (size_t)(b * kM + j) * 3;
    _Float16* fr = frow + r * kCf;
    fr[64] = (_Float16)(pj[0] - pi[0]);
    fr[65] = (_Float16)(pj[1] - pi[1]);
    fr[66] = (_Float16)(pj[2] - pi[2]);
    #pragma unroll
    for (int t = 67; t < kCf; ++t) fr[t] = (_Float16)0.f;
  }
  wait_asynccnt0();                                 // async LDS writes visible
  __syncthreads();

  const int hx = lane >> 4, mrow = lane & 15, ncol = lane & 15;
  #pragma unroll
  for (int t = 0; t < 4; ++t) acc[t] = vzero8();
  #pragma unroll
  for (int kb = 0; kb < 3; ++kb) {
    v16h a = ld_fragA(frow + mrow * kCf, kb, hx);
    #pragma unroll
    for (int t = 0; t < 4; ++t) {
      v16h bf = ld_fragB(w1t + (t * 16 + ncol) * kCf, kb, hx);
      acc[t] = __builtin_amdgcn_wmma_f32_16x16x32_f16(false, a, false, bf,
                                                      (short)0, acc[t], false, false);
    }
  }
}

// ------------------------------------------------------- pass 1: BN stats ----
__global__ __launch_bounds__(256) void stats_kernel(const float* __restrict__ p,
                                                    const _Float16* __restrict__ xh,
                                                    const int* __restrict__ idx,
                                                    const _Float16* __restrict__ w1t,
                                                    float* __restrict__ gsum,
                                                    float* __restrict__ gsq) {
  __shared__ __attribute__((aligned(16))) _Float16 feat[8 * 16 * kCf];
  __shared__ float bsum[64], bsq[64];
  if (threadIdx.x < 64) { bsum[threadIdx.x] = 0.f; bsq[threadIdx.x] = 0.f; }
  if (threadIdx.x == 0) __builtin_prefetch(w1t, 0, 0);

  const int wave = threadIdx.x >> 5, lane = threadIdx.x & 31;
  const int point = blockIdx.x * 8 + wave;
  v8f acc[4];
  stage_and_gemm1(p, xh, idx, w1t, feat + wave * (16 * kCf), point, lane, acc);

  const int ncol = lane & 15;
  #pragma unroll
  for (int t = 0; t < 4; ++t) {
    float s = 0.f, q = 0.f;
    #pragma unroll
    for (int rr = 0; rr < 8; ++rr) { float v = acc[t][rr]; s += v; q += v * v; }
    s += __shfl_xor(s, 16);                          // combine M=0..7 with M=8..15
    q += __shfl_xor(q, 16);
    if (lane < 16) {
      atomicAdd(&bsum[t * 16 + ncol], s);
      atomicAdd(&bsq[t * 16 + ncol], q);
    }
  }
  __syncthreads();
  if (threadIdx.x < 64) atomicAdd(&gsum[threadIdx.x], bsum[threadIdx.x]);
  else if (threadIdx.x < 128) atomicAdd(&gsq[threadIdx.x - 64], bsq[threadIdx.x - 64]);
}

__global__ void finalize_kernel(const float* __restrict__ gamma,
                                const float* __restrict__ beta,
                                float* __restrict__ stat) {
  int d = threadIdx.x;                               // 64 threads
  float n = (float)((long long)kPts * kK);
  float mean = stat[d] / n;
  float var  = stat[64 + d] / n - mean * mean;
  float sc   = gamma[d] * rsqrtf(var + kEps);
  stat[128 + d] = sc;                                // scale
  stat[192 + d] = beta[d] - mean * sc;               // shift
}

// --------------------------- pass 2: GEMM1 + BN + ReLU + GEMM2 + k-maxpool ---
__global__ __launch_bounds__(256) void output_kernel(const float* __restrict__ p,
                                                     const _Float16* __restrict__ xh,
                                                     const int* __restrict__ idx,
                                                     const _Float16* __restrict__ w1t,
                                                     const _Float16* __restrict__ w2t,
                                                     const float* __restrict__ scale,
                                                     const float* __restrict__ shift,
                                                     const float* __restrict__ b2,
                                                     float* __restrict__ out) {
  __shared__ __attribute__((aligned(16))) _Float16 feat[8 * 16 * kCf];
  if (threadIdx.x == 0) __builtin_prefetch(w2t, 0, 0);
  const int wave = threadIdx.x >> 5, lane = threadIdx.x & 31;
  const int point = blockIdx.x * 8 + wave;
  _Float16* frow = feat + wave * (16 * kCf);

  v8f acc[4];
  stage_and_gemm1(p, xh, idx, w1t, frow, point, lane, acc);

  const int hx = lane >> 4, ncol = lane & 15, mrow = lane & 15;
  __syncthreads();

  // BN affine + ReLU, write h back to LDS as f16 [16 rows][64] (stride 64)
  #pragma unroll
  for (int t = 0; t < 4; ++t) {
    int d = t * 16 + ncol;
    float sc = scale[d], sh = shift[d];
    #pragma unroll
    for (int rr = 0; rr < 8; ++rr) {
      float v = fmaxf(acc[t][rr] * sc + sh, 0.f);
      frow[(rr + 8 * hx) * 64 + d] = (_Float16)v;
    }
  }
  __syncthreads();

  // GEMM2: D(16x128) = h(16x64) * w2t  -> 2 K-chunks x 8 N-tiles
  v8f acc2[8];
  #pragma unroll
  for (int nt = 0; nt < 8; ++nt) acc2[nt] = vzero8();
  #pragma unroll
  for (int kb = 0; kb < 2; ++kb) {
    v16h a = ld_fragA(frow + mrow * 64, kb, hx);
    #pragma unroll
    for (int nt = 0; nt < 8; ++nt) {
      v16h bf = ld_fragB(w2t + (nt * 16 + ncol) * 64, kb, hx);
      acc2[nt] = __builtin_amdgcn_wmma_f32_16x16x32_f16(false, a, false, bf,
                                                        (short)0, acc2[nt], false, false);
    }
  }

  // max over 16 neighbor rows, add bias (max(x)+b == max(x+b), b const per ch)
  #pragma unroll
  for (int nt = 0; nt < 8; ++nt) {
    float mx = acc2[nt][0];
    #pragma unroll
    for (int rr = 1; rr < 8; ++rr) mx = fmaxf(mx, acc2[nt][rr]);
    mx = fmaxf(mx, __shfl_xor(mx, 16));
    if (lane < 16) {
      int ch = nt * 16 + ncol;
      out[(size_t)point * kCout + ch] = mx + b2[ch];
    }
  }
}

// ------------------------------------------------------------------ launch ---
extern "C" void kernel_launch(void* const* d_in, const int* in_sizes, int n_in,
                              void* d_out, int out_size, void* d_ws, size_t ws_size,
                              hipStream_t stream) {
  (void)in_sizes; (void)n_in; (void)out_size; (void)ws_size;
  const float* p     = (const float*)d_in[0];
  const float* x     = (const float*)d_in[1];
  const float* w1    = (const float*)d_in[2];
  const float* gamma = (const float*)d_in[3];
  const float* beta  = (const float*)d_in[4];
  const float* w2    = (const float*)d_in[5];
  const float* b2    = (const float*)d_in[6];
  float* out = (float*)d_out;

  char* ws = (char*)d_ws;
  int*      idx  = (int*)(ws + OFF_IDX);
  _Float16* xh   = (_Float16*)(ws + OFF_XH);
  _Float16* w1t  = (_Float16*)(ws + OFF_W1T);
  _Float16* w2t  = (_Float16*)(ws + OFF_W2T);
  float*    stat = (float*)(ws + OFF_STAT);

  prep_x_kernel<<<kPts * kCin / 256, 256, 0, stream>>>(x, xh);
  prep_w_kernel<<<(64 * 96 + 128 * 64 + 128 + 255) / 256, 256, 0, stream>>>(w1, w2, w1t, w2t, stat);
  knn_kernel<<<dim3(kM / 256, kB), 256, 0, stream>>>(p, idx);
  stats_kernel<<<kPts / 8, 256, 0, stream>>>(p, xh, idx, w1t, stat, stat + 64);
  finalize_kernel<<<1, 64, 0, stream>>>(gamma, beta, stat);
  output_kernel<<<kPts / 8, 256, 0, stream>>>(p, xh, idx, w1t, w2t, stat + 128, stat + 192, b2, out);
}
